// FlexMessagePassingConvolution_1778116461346
// MI455X (gfx1250) — matrix operand order repaired and checked by reading, not compiled, not run on hardware
//
#include <hip/hip_runtime.h>
#include <hip/hip_bf16.h>

typedef __attribute__((ext_vector_type(16))) _Float16 v16h;
typedef __attribute__((ext_vector_type(8)))  _Float16 v8h;
typedef __attribute__((ext_vector_type(8)))  float    v8f;

#define S_CH 64
#define V_CH 32
#define R_CH 32
#define MLPW 128
#define ROWSTRIDE 200          // 192 K-cols padded to 200 halfs (400B) to avoid LDS bank conflicts
#define WAVES 8                // 8 waves * 16 edges = 128 edges per block

// packed weight offsets (in halfs) inside d_ws
#define W1P_OFF 0              // [6][8][32][16]  = 24576 halfs (K padded 161->192)
#define W2P_OFF 24576          // [4][8][32][16]  = 16384 halfs
#define W3P_OFF 40960          // [4][8][32][16]  = 16384 halfs
#define PACK_TOTAL 57344

// ---------------- zero output ----------------
__global__ void zero_out_kernel(float* __restrict__ p, int n) {
    int i = blockIdx.x * 256 + threadIdx.x;
    if (i < n) p[i] = 0.f;
}

// ---------------- pack weights to f16 WMMA-B fragment layout ----------------
// B fragment (16-bit, 32x16): lane L holds column n = L&15; half j = B[(j>>3)*16 + (L>>4)*8 + (j&7)][n]
// storage: [kt][nt][lane][16 halfs]  -> each lane's v16h is 32 contiguous bytes
__global__ void pack_weights_kernel(const float* __restrict__ W1,
                                    const float* __restrict__ W2,
                                    const float* __restrict__ W3,
                                    _Float16* __restrict__ wp) {
    int idx = blockIdx.x * 256 + threadIdx.x;
    if (idx >= PACK_TOTAL) return;
    const float* W; int K; int off;
    if (idx < W2P_OFF)      { W = W1; K = 161; off = idx; }
    else if (idx < W3P_OFF) { W = W2; K = 128; off = idx - W2P_OFF; }
    else                    { W = W3; K = 128; off = idx - W3P_OFF; }
    int j    = off & 15;
    int lane = (off >> 4) & 31;
    int nt   = (off >> 9) & 7;
    int kt   = off >> 12;
    int k = kt * 32 + (j >> 3) * 16 + (lane >> 4) * 8 + (j & 7);
    int n = nt * 16 + (lane & 15);
    float v = (k < K) ? W[k * MLPW + n] : 0.f;
    wp[idx] = (_Float16)v;
}

// ---------------- one MLP layer on a 16-edge tile via WMMA ----------------
template<int KT, bool ACT>
__device__ __forceinline__ void mlp_layer(_Float16 (* __restrict__ Aw)[ROWSTRIDE],
                                          const _Float16* __restrict__ Wp,
                                          const float* __restrict__ bias,
                                          int lane) {
    const int m = lane & 15;
    const int h = lane >> 4;
    v8f acc[8] = {};
#pragma unroll
    for (int kt = 0; kt < KT; ++kt) {
        // 16-bit A 16x32 layout: lanes h=0 hold K=[kt*32+0..7 | +16..23], h=1 hold [+8..15 | +24..31]
        v8h lo = *(const v8h*)&Aw[m][kt * 32 + h * 8];
        v8h hi = *(const v8h*)&Aw[m][kt * 32 + 16 + h * 8];
        v16h a;
#pragma unroll
        for (int q = 0; q < 8; ++q) { a[q] = lo[q]; a[q + 8] = hi[q]; }
#pragma unroll
        for (int nt = 0; nt < 8; ++nt) {
            v16h b = *(const v16h*)(Wp + (((kt * 8 + nt) * 32) + lane) * 16);
            acc[nt] = __builtin_amdgcn_wmma_f32_16x16x32_f16(
                false, a, false, b, (short)0, acc[nt], false, false);
        }
    }
    __syncthreads();
    // D layout: VGPR r, lane L -> row m = r + 8*(L>>4), col n = L&15 (per 16-col tile)
#pragma unroll
    for (int nt = 0; nt < 8; ++nt) {
        int n = nt * 16 + (lane & 15);
        float bn = bias[n];
#pragma unroll
        for (int r = 0; r < 8; ++r) {
            float x = acc[nt][r] + bn;
            if (ACT) x = x / (1.f + __expf(-x));     // silu
            Aw[r + 8 * h][n] = (_Float16)x;
        }
    }
    __syncthreads();
}

// ---------------- fused edge kernel ----------------
__global__ __launch_bounds__(256)
void edge_mp_kernel(const float* __restrict__ node_scalars,   // [N,64]
                    const float* __restrict__ node_vectors,   // [N,32,3]
                    const float* __restrict__ edge_vec,       // [E,3]
                    const float* __restrict__ edge_feats,     // [E,32]
                    const float* __restrict__ lengths,        // [E,1]
                    const int*   __restrict__ senders,
                    const int*   __restrict__ receivers,
                    const float* __restrict__ b1,
                    const float* __restrict__ b2,
                    const float* __restrict__ b3,
                    const _Float16* __restrict__ wp,          // packed weights
                    float* __restrict__ out,                  // [N,320]
                    int E) {
    __shared__ _Float16 buf[WAVES][16][ROWSTRIDE];            // 51200 B
    const int lane = threadIdx.x & 31;
    const int wave = threadIdx.x >> 5;
    const int e0 = blockIdx.x * (WAVES * 16) + wave * 16;
    _Float16 (* __restrict__ Aw)[ROWSTRIDE] = buf[wave];

    // ---- gather mlp_in = [s_snd(64) | s_rcv(64) | edge_feats(32) | len(1) | pad->192] ----
    for (int i = 0; i < 16; ++i) {
        int e = e0 + i;
        int ec = (e < E) ? e : 0;
        int snd = senders[ec];
        int rcv = receivers[ec];
        Aw[i][lane]          = (_Float16)node_scalars[snd * S_CH + lane];
        Aw[i][lane + 32]     = (_Float16)node_scalars[snd * S_CH + lane + 32];
        Aw[i][64 + lane]     = (_Float16)node_scalars[rcv * S_CH + lane];
        Aw[i][96 + lane]     = (_Float16)node_scalars[rcv * S_CH + lane + 32];
        Aw[i][128 + lane]    = (_Float16)edge_feats[ec * R_CH + lane];
        if (lane == 0) Aw[i][160] = (_Float16)lengths[ec];
        if (lane < 31) Aw[i][161 + lane] = (_Float16)0.f;     // zero-pad K 161..191
    }
    __syncthreads();

    // ---- MLP: 161(->192) -> 128 silu -> 128 silu -> 128 gates ----
    mlp_layer<6, true >(Aw, wp + W1P_OFF, b1, lane);
    mlp_layer<4, true >(Aw, wp + W2P_OFF, b2, lane);
    mlp_layer<4, false>(Aw, wp + W3P_OFF, b3, lane);          // mix now in Aw[i][0..127]

    // ---- tensor product + gated scatter-add (already divided by AVG=10) ----
    const float K_DOT = 0.1f / 1.7320508075688772f;           // (1/sqrt3)/AVG
    const float K_OUT = 0.1f;                                 // 1/AVG
    const float K_CRS = 0.1f / 1.4142135623730951f;           // (1/sqrt2)/AVG
    const int c = lane;                                       // 0..31
    for (int i = 0; i < 16; ++i) {
        int e = e0 + i;
        if (e >= E) break;
        int snd = senders[e];
        int rcv = receivers[e];
        float ev0 = edge_vec[e * 3 + 0];
        float ev1 = edge_vec[e * 3 + 1];
        float ev2 = edge_vec[e * 3 + 2];
        float* outp = out + (long)rcv * 320;

        // vector channel c of sender
        const float* vp = node_vectors + snd * (V_CH * 3) + c * 3;
        float vx = vp[0], vy = vp[1], vz = vp[2];

        // m0: (1o x 1o -> 0e) gated
        float mix0 = (float)Aw[i][c];
        float dot = vx * ev0 + vy * ev1 + vz * ev2;
        atomicAdd(outp + c, dot * mix0 * K_DOT);

        // m1b: cross(v, ev)/sqrt2 gated -> channels 64..95 of out1
        float mixb = (float)Aw[i][96 + c] * K_CRS;
        float cx = vy * ev2 - vz * ev1;
        float cy = vz * ev0 - vx * ev2;
        float cz = vx * ev1 - vy * ev0;
        float* pb = outp + 32 + (64 + c) * 3;
        atomicAdd(pb + 0, cx * mixb);
        atomicAdd(pb + 1, cy * mixb);
        atomicAdd(pb + 2, cz * mixb);

        // m1a: s_snd[j] * ev gated -> channels 0..63 of out1 (lane handles j=c, c+32)
#pragma unroll
        for (int t = 0; t < 2; ++t) {
            int j = c + t * 32;
            float s  = node_scalars[snd * S_CH + j];
            float ma = (float)Aw[i][32 + j] * s * K_OUT;
            float* pa = outp + 32 + j * 3;
            atomicAdd(pa + 0, ma * ev0);
            atomicAdd(pa + 1, ma * ev1);
            atomicAdd(pa + 2, ma * ev2);
        }
    }
}

extern "C" void kernel_launch(void* const* d_in, const int* in_sizes, int n_in,
                              void* d_out, int out_size, void* d_ws, size_t ws_size,
                              hipStream_t stream) {
    const float* node_scalars = (const float*)d_in[0];
    const float* node_vectors = (const float*)d_in[1];
    const float* edge_vec     = (const float*)d_in[2];
    const float* edge_feats   = (const float*)d_in[3];
    const float* lengths      = (const float*)d_in[4];
    const int*   senders      = (const int*)  d_in[5];
    const int*   receivers    = (const int*)  d_in[6];
    const float* W1 = (const float*)d_in[7];
    const float* b1 = (const float*)d_in[8];
    const float* W2 = (const float*)d_in[9];
    const float* b2 = (const float*)d_in[10];
    const float* W3 = (const float*)d_in[11];
    const float* b3 = (const float*)d_in[12];
    float* out = (float*)d_out;
    _Float16* wp = (_Float16*)d_ws;
    const int E = in_sizes[5];

    zero_out_kernel<<<(out_size + 255) / 256, 256, 0, stream>>>(out, out_size);
    pack_weights_kernel<<<PACK_TOTAL / 256, 256, 0, stream>>>(W1, W2, W3, wp);
    int blocks = (E + WAVES * 16 - 1) / (WAVES * 16);
    edge_mp_kernel<<<blocks, 256, 0, stream>>>(
        node_scalars, node_vectors, edge_vec, edge_feats, lengths,
        senders, receivers, b1, b2, b3, wp, out, E);
}
